// FusionMamba_83150566850926
// MI455X (gfx1250) — matrix operand order repaired and checked by reading, not compiled, not run on hardware
//
#include <hip/hip_runtime.h>
#include <hip/hip_bf16.h>
#include <cstddef>

typedef __attribute__((ext_vector_type(16))) _Float16 v16h;
typedef __attribute__((ext_vector_type(8)))  float    v8f;

#define FM_B 4
#define FM_L 4096
#define FM_D 64
#define FM_M (FM_B*FM_L)      /* 16384 rows */
#define FM_NS 16              /* d_state */
#define FM_CHUNK 64
#define FM_NCH 64

// ---------------------------------------------------------------- device math
__device__ __forceinline__ float fm_silu(float x){ return x/(1.f+__expf(-x)); }
__device__ __forceinline__ float fm_sigmoid(float x){ return 1.f/(1.f+__expf(-x)); }
__device__ __forceinline__ float fm_gelu(float x){
  float u = 0.7978845608028654f*(x + 0.044715f*x*x*x);
  return 0.5f*x*(1.f+tanhf(u));
}

// ---------------------------------------------------------------- WMMA GEMM
// out[M,N] = act[M,K] @ W^T   (W is [N,K] row-major, as in x @ W.T)
// Stage 1: raw W panel is DMA'd into LDS with GLOBAL_LOAD_ASYNC_TO_LDS_B128
//          (per-lane global->LDS, no VGPR round trip, ASYNCcnt-tracked).
// Stage 2: LDS->LDS swizzle into a B lane layout with contiguous k = 16*(l/16)+s (f16),
//          so each lane later reads a contiguous v16h with ds_load_b128 x2.
template<int KT, int NT, int ACT>
__global__ void __launch_bounds__(256)
k_gemm(const float* __restrict__ A, int lda,
       const float* __restrict__ W, int N,
       const float* __restrict__ bias,
       float* __restrict__ out, int ldo)
{
  __shared__ float    ldsRaw[KT*NT*512];   // raw f32 W rows [n][k]
  __shared__ _Float16 ldsB  [KT*NT*512];   // swizzled f16 B fragments
  const int K = KT*32;

  // ---- stage 1: async global->LDS copy of the weight panel (b128 per lane)
  {
    const int totalVec = (N*K) >> 2;       // float4 count (K mult of 32 => exact)
    for (int i = threadIdx.x; i < totalVec; i += 256){
      unsigned lofs = (unsigned)(size_t)(void*)&ldsRaw[i*4];
      unsigned long long ga = (unsigned long long)(const void*)(W + (size_t)i*4);
      asm volatile("global_load_async_to_lds_b128 %0, %1, off"
                   :: "v"(lofs), "v"(ga) : "memory");
    }
    asm volatile("s_wait_asynccnt 0x0" ::: "memory");
    __syncthreads();
  }

  // ---- stage 2: swizzle f32 raw -> f16 B layout
  //   region (kt,nt): lane l, half h -> n = nt*16+(l&15), k = kt*32+(l>>4)*16+h
  for (int idx = threadIdx.x; idx < KT*NT*512; idx += 256){
    int region = idx >> 9;
    int within = idx & 511;
    int ln = within >> 4;
    int h  = within & 15;
    int kt = region / NT;
    int nt = region % NT;
    int k  = kt*32 + ((ln>>4)<<4) + h;
    int n  = nt*16 + (ln&15);
    float w = (n < N) ? ldsRaw[n*K + k] : 0.f;
    ldsB[idx] = (_Float16)w;
  }
  __syncthreads();

  const int lane = threadIdx.x & 31;
  const int wid  = threadIdx.x >> 5;
  const int rowBase = blockIdx.x*128 + wid*16;
  const int mlane = lane & 15;
  const int hsel  = lane >> 4;

  v8f acc[NT];
  #pragma unroll
  for (int nt = 0; nt < NT; ++nt)
    #pragma unroll
    for (int j = 0; j < 8; ++j) acc[nt][j] = 0.f;

  #pragma unroll
  for (int kt = 0; kt < KT; ++kt){
    // A fragment: contiguous k order (same as ldsB), VGPR v holds K = hsel*16+v*2 (+1)
    const float* arow = A + (size_t)(rowBase + mlane)*lda + kt*32;
    v16h a;
    #pragma unroll
    for (int v = 0; v < 8; ++v){
      int kb = hsel*16 + (v << 1);
      a[2*v]   = (_Float16)arow[kb];
      a[2*v+1] = (_Float16)arow[kb+1];
    }
    #pragma unroll
    for (int nt = 0; nt < NT; ++nt){
      v16h bf = *(const v16h*)&ldsB[((kt*NT + nt) << 9) + (lane << 4)];
      acc[nt] = __builtin_amdgcn_wmma_f32_16x16x32_f16(
          false, a, false, bf, (short)0, acc[nt], false, false);
    }
  }

  // D layout: lane 0-15 -> N=lane, M=0..7 in v0..7 ; lane 16-31 -> M=8..15
  const int ncol  = lane & 15;
  const int mrow0 = hsel*8;
  #pragma unroll
  for (int nt = 0; nt < NT; ++nt){
    int n = nt*16 + ncol;
    if (n >= N) continue;
    float bb = bias ? bias[n] : 0.f;
    #pragma unroll
    for (int v = 0; v < 8; ++v){
      float val = acc[nt][v] + bb;
      if (ACT == 1) val = fm_silu(val);
      else if (ACT == 2) val = fm_sigmoid(val);
      else if (ACT == 3) val = fm_gelu(val);
      out[(size_t)(rowBase + mrow0 + v)*ldo + n] = val;
    }
  }
}

// ---------------------------------------------------------------- elementwise
__global__ void k_convsilu(const float* __restrict__ x, int ld,
                           const float* __restrict__ w,
                           const float* __restrict__ bias,
                           float* __restrict__ out)
{
  int tid = blockIdx.x*blockDim.x + threadIdx.x;   // M*64
  int dch = tid & 63;
  int l   = (tid >> 6) & (FM_L - 1);
  int b   = tid >> 18;
  float acc = bias[dch];
  #pragma unroll
  for (int j = 0; j < 4; ++j){
    int li = l - 3 + j;
    if (li >= 0) acc += w[dch*4 + j] * x[(size_t)(b*FM_L + li)*ld + dch];
  }
  out[tid] = fm_silu(acc);
}

__global__ void k_reverse(const float* __restrict__ in, int ld, float* __restrict__ out)
{
  int tid = blockIdx.x*blockDim.x + threadIdx.x;
  int dch = tid & 63;
  int l   = (tid >> 6) & (FM_L - 1);
  int b   = tid >> 18;
  out[tid] = in[(size_t)(b*FM_L + (FM_L - 1 - l))*ld + dch];
}

__global__ void k_avgrev(const float* __restrict__ yf, const float* __restrict__ yb,
                         float* __restrict__ out)
{
  int tid = blockIdx.x*blockDim.x + threadIdx.x;
  int dch = tid & 63;
  int l   = (tid >> 6) & (FM_L - 1);
  int b   = tid >> 18;
  out[tid] = 0.5f*(yf[tid] + yb[(size_t)(b*FM_L + (FM_L - 1 - l))*64 + dch]);
}

__global__ void k_dtsp(const float* __restrict__ dbc,
                       const float* __restrict__ dtw,
                       const float* __restrict__ dtb,
                       float* __restrict__ out)
{
  int tid = blockIdx.x*blockDim.x + threadIdx.x;   // M*64
  int dch = tid & 63;
  int m   = tid >> 6;
  const float* r = dbc + (size_t)m*48;
  float s = dtb[dch];
  #pragma unroll
  for (int j = 0; j < 4; ++j) s += r[j]*dtw[dch*4 + j];
  out[tid] = (s > 20.f) ? s : log1pf(__expf(s));
}

// ---------------------------------------------------------------- SSM scan
// phase 1: per-(b,d,chunk) sequential recurrence -> carry (aProd[16], h[16])
__global__ void __launch_bounds__(256)
k_scan_chunk(const float* __restrict__ xc, const float* __restrict__ dt,
             const float* __restrict__ dbc, const float* __restrict__ A_log,
             float* __restrict__ carry)
{
  int tid = blockIdx.x*256 + threadIdx.x;          // B*64*NCH = 16384 lanes
  int dch = tid & 63;
  int ch  = (tid >> 6) & (FM_NCH - 1);
  int b   = tid >> 12;
  float A[FM_NS], aP[FM_NS], h[FM_NS];
  #pragma unroll
  for (int n = 0; n < FM_NS; ++n){
    A[n] = -__expf(A_log[dch*FM_NS + n]); aP[n] = 1.f; h[n] = 0.f;
  }
  int l0 = ch*FM_CHUNK;
  for (int t = 0; t < FM_CHUNK; ++t){
    size_t row = (size_t)b*FM_L + l0 + t;
    float dtv = dt[row*64 + dch];
    float xv  = xc[row*64 + dch];
    __builtin_prefetch(dt + (row+4)*64 + dch, 0, 0);
    float dtx = dtv*xv;
    const float* Bp = dbc + row*48 + 4;
    #pragma unroll
    for (int n = 0; n < FM_NS; ++n){
      float da = __expf(dtv*A[n]);
      aP[n] *= da;
      h[n] = da*h[n] + dtx*Bp[n];
    }
  }
  size_t base = (size_t)((b*FM_NCH + ch)*FM_NS)*128 + dch;
  #pragma unroll
  for (int n = 0; n < FM_NS; ++n){
    carry[base + (size_t)n*128]      = aP[n];
    carry[base + (size_t)n*128 + 64] = h[n];
  }
}

// phase 2: exclusive scan over chunk carries per (b,d,n)
__global__ void k_scan_carry(const float* __restrict__ carry, float* __restrict__ cin)
{
  int tid = blockIdx.x*blockDim.x + threadIdx.x;   // 4096
  int dch = tid & 63;
  int n   = (tid >> 6) & 15;
  int b   = tid >> 10;
  float h = 0.f;
  for (int ch = 0; ch < FM_NCH; ++ch){
    size_t ci = (size_t)(b*FM_NCH + ch)*FM_NS + n;
    cin[ci*64 + dch] = h;
    float a  = carry[ci*128 + dch];
    float bs = carry[ci*128 + 64 + dch];
    h = a*h + bs;
  }
}

// phase 3: replay with carry-in, emit y = h.C + x*D
__global__ void __launch_bounds__(256)
k_scan_apply(const float* __restrict__ xc, const float* __restrict__ dt,
             const float* __restrict__ dbc, const float* __restrict__ A_log,
             const float* __restrict__ Dp, const float* __restrict__ cin,
             float* __restrict__ y)
{
  int tid = blockIdx.x*256 + threadIdx.x;
  int dch = tid & 63;
  int ch  = (tid >> 6) & (FM_NCH - 1);
  int b   = tid >> 12;
  float A[FM_NS], h[FM_NS];
  #pragma unroll
  for (int n = 0; n < FM_NS; ++n){
    A[n] = -__expf(A_log[dch*FM_NS + n]);
    h[n] = cin[((size_t)(b*FM_NCH + ch)*FM_NS + n)*64 + dch];
  }
  float Dv = Dp[dch];
  int l0 = ch*FM_CHUNK;
  for (int t = 0; t < FM_CHUNK; ++t){
    size_t row = (size_t)b*FM_L + l0 + t;
    float dtv = dt[row*64 + dch];
    float xv  = xc[row*64 + dch];
    __builtin_prefetch(xc + (row+4)*64 + dch, 0, 0);
    float dtx = dtv*xv;
    const float* Bp = dbc + row*48 + 4;
    const float* Cp = dbc + row*48 + 20;
    float yv = xv*Dv;
    #pragma unroll
    for (int n = 0; n < FM_NS; ++n){
      float da = __expf(dtv*A[n]);
      h[n] = da*h[n] + dtx*Bp[n];
      yv += h[n]*Cp[n];
    }
    y[row*64 + dch] = yv;
  }
}

// ---------------------------------------------------------------- norms (wave/token)
__global__ void k_layernorm(const float* __restrict__ x, const float* __restrict__ w,
                            const float* __restrict__ b, float* __restrict__ out)
{
  int wid = threadIdx.x >> 5, lane = threadIdx.x & 31;
  int t = blockIdx.x*8 + wid;
  float2 v = ((const float2*)(x + (size_t)t*64))[lane];
  float s = v.x + v.y;
  for (int m = 16; m > 0; m >>= 1) s += __shfl_xor(s, m, 32);
  float mean = s*(1.f/64.f);
  float dx = v.x - mean, dy = v.y - mean;
  float q = dx*dx + dy*dy;
  for (int m = 16; m > 0; m >>= 1) q += __shfl_xor(q, m, 32);
  float rstd = rsqrtf(q*(1.f/64.f) + 1e-5f);
  int c = lane*2;
  float2 o; o.x = dx*rstd*w[c] + b[c]; o.y = dy*rstd*w[c+1] + b[c+1];
  ((float2*)(out + (size_t)t*64))[lane] = o;
}

__global__ void k_rmsgate(const float* __restrict__ y, const float* __restrict__ w,
                          const float* __restrict__ z, int ldz, float* __restrict__ out)
{
  int wid = threadIdx.x >> 5, lane = threadIdx.x & 31;
  int t = blockIdx.x*8 + wid;
  float2 v = ((const float2*)(y + (size_t)t*64))[lane];
  float q = v.x*v.x + v.y*v.y;
  for (int m = 16; m > 0; m >>= 1) q += __shfl_xor(q, m, 32);
  float r = rsqrtf(q*(1.f/64.f) + 1e-5f);
  int c = lane*2;
  const float* zr = z + (size_t)t*ldz;
  float2 o;
  o.x = v.x*r*w[c]   * fm_silu(zr[c]);
  o.y = v.y*r*w[c+1] * fm_silu(zr[c+1]);
  ((float2*)(out + (size_t)t*64))[lane] = o;
}

// ---------------------------------------------------------------- misc elementwise
__global__ void k_concat(const float* __restrict__ a, const float* __restrict__ b,
                         float* __restrict__ cat)
{
  int tid = blockIdx.x*blockDim.x + threadIdx.x;   // M*128
  int c = tid & 127;
  int m = tid >> 7;
  cat[tid] = (c < 64) ? a[(size_t)m*64 + c] : b[(size_t)m*64 + c - 64];
}

__global__ void k_crossgate(const float* __restrict__ mo, const float* __restrict__ wgt,
                            const float* __restrict__ n0, const float* __restrict__ skip,
                            float* __restrict__ out)
{
  int tid = blockIdx.x*blockDim.x + threadIdx.x;
  float g = wgt[tid];
  out[tid] = mo[tid]*g + n0[tid]*(1.f - g) + skip[tid];
}

__global__ void k_add(const float* __restrict__ a, const float* __restrict__ b,
                      float* __restrict__ out)
{
  int tid = blockIdx.x*blockDim.x + threadIdx.x;
  out[tid] = a[tid] + b[tid];
}

__global__ void k_fuse(const float* __restrict__ spa, const float* __restrict__ spe,
                       const float* __restrict__ alpha, const float* __restrict__ beta,
                       float* __restrict__ out)
{
  int tid = blockIdx.x*blockDim.x + threadIdx.x;
  float a = alpha[0], b = beta[0];
  float inv = 1.f/(a + b + 1e-8f);
  out[tid] = (a*spa[tid] + b*spe[tid])*inv;
}

// ---------------------------------------------------------------- transposes
__global__ void k_tok(const float* __restrict__ in, float* __restrict__ out)
{
  __shared__ float tile[32][33];
  int b = blockIdx.z, c0 = blockIdx.y*32, l0 = blockIdx.x*32;
  int tx = threadIdx.x, ty = threadIdx.y;
  #pragma unroll
  for (int i = 0; i < 32; i += 8)
    tile[ty+i][tx] = in[(size_t)(b*64 + c0+ty+i)*FM_L + l0+tx];
  __syncthreads();
  #pragma unroll
  for (int i = 0; i < 32; i += 8)
    out[(size_t)(b*FM_L + l0+ty+i)*64 + c0+tx] = tile[tx][ty+i];
}

__global__ void k_final(const float* __restrict__ f, const float* __restrict__ x1,
                        const float* __restrict__ x2, float* __restrict__ out)
{
  __shared__ float t0[32][33], t1[32][33], t2[32][33];
  int b = blockIdx.z, c0 = blockIdx.y*32, l0 = blockIdx.x*32;
  int tx = threadIdx.x, ty = threadIdx.y;
  #pragma unroll
  for (int i = 0; i < 32; i += 8){
    size_t idx = (size_t)(b*FM_L + l0+ty+i)*64 + c0+tx;
    t0[ty+i][tx] = f[idx];
    t1[ty+i][tx] = x1[idx];
    t2[ty+i][tx] = x2[idx];
  }
  __syncthreads();
  const size_t half = (size_t)FM_B*64*FM_L;
  #pragma unroll
  for (int i = 0; i < 32; i += 8){
    size_t o = (size_t)(b*64 + c0+ty+i)*FM_L + l0+tx;
    float fv = t0[tx][ty+i];
    out[o]        = (t1[tx][ty+i] + fv)*0.5f;
    out[half + o] = (t2[tx][ty+i] + fv)*0.5f;
  }
}

// ================================================================ host side
namespace {

struct Arena {
  char* base; size_t off;
  void* alloc(size_t n){ n = (n + 255) & ~(size_t)255; void* p = base + off; off += n; return p; }
};

constexpr size_t S64B  = (size_t)FM_M*64*sizeof(float);
constexpr size_t S128B = (size_t)FM_M*128*sizeof(float);
constexpr size_t S48B  = (size_t)FM_M*48*sizeof(float);
constexpr size_t CARRYB = (size_t)FM_B*64*FM_NCH*32*sizeof(float);
constexpr size_t CINB   = (size_t)FM_B*64*FM_NCH*16*sizeof(float);
constexpr int G64 = (FM_M*64)/256;    // 4096
constexpr int G128 = (FM_M*128)/256;  // 8192
constexpr int GLN = FM_M/8;           // 2048
constexpr int GSCAN = (FM_B*64*FM_NCH)/256;  // 64

struct MambaP {
  const float *A_log,*A_log_b,*D,*D_b,*conv_b,*conv_b_b,*conv_w,*conv_w_b;
  const float *dt_b,*dt_b_b,*dt_w,*dt_w_b;
  const float *econv_b,*econv_b_b,*econv_w,*econv_w_b,*extra_w;
  const float *in_w,*out_norm_w,*out_w,*x_w,*x_w_b;
  bool cross;
};
struct CrossP {
  const float *cw_b,*cw_w,*ln0_b,*ln0_w,*ln1_b,*ln1_w;
  MambaP m;
};

MambaP readMamba(void* const* d_in, int& pi, bool cross){
  auto nx = [&]{ return (const float*)d_in[pi++]; };
  MambaP m{};
  m.A_log = nx(); m.A_log_b = nx(); m.D = nx(); m.D_b = nx();
  m.conv_b = nx(); m.conv_b_b = nx(); m.conv_w = nx(); m.conv_w_b = nx();
  m.dt_b = nx(); m.dt_b_b = nx(); m.dt_w = nx(); m.dt_w_b = nx();
  if (cross){ m.econv_b = nx(); m.econv_b_b = nx(); m.econv_w = nx(); m.econv_w_b = nx(); m.extra_w = nx(); }
  m.in_w = nx(); m.out_norm_w = nx(); m.out_w = nx(); m.x_w = nx(); m.x_w_b = nx();
  m.cross = cross;
  return m;
}
CrossP readCross(void* const* d_in, int& pi){
  auto nx = [&]{ return (const float*)d_in[pi++]; };
  CrossP c{};
  c.cw_b = nx(); c.cw_w = nx(); c.ln0_b = nx(); c.ln0_w = nx(); c.ln1_b = nx(); c.ln1_w = nx();
  c.m = readMamba(d_in, pi, true);
  return c;
}

void run_ssm(hipStream_t s, Arena& ar, const float* src, const float* xc,
             const float* x_w, const float* dt_w, const float* dt_b,
             const float* A_log, const float* Dp, float* y)
{
  size_t save = ar.off;
  float* dbc = (float*)ar.alloc(S48B);
  k_gemm<2,3,0><<<128,256,0,s>>>(src, 64, x_w, 36, nullptr, dbc, 48);
  float* dt = (float*)ar.alloc(S64B);
  k_dtsp<<<G64,256,0,s>>>(dbc, dt_w, dt_b, dt);
  float* carry = (float*)ar.alloc(CARRYB);
  float* cin   = (float*)ar.alloc(CINB);
  k_scan_chunk<<<GSCAN,256,0,s>>>(xc, dt, dbc, A_log, carry);
  k_scan_carry<<<16,256,0,s>>>(carry, cin);
  k_scan_apply<<<GSCAN,256,0,s>>>(xc, dt, dbc, A_log, Dp, cin, y);
  ar.off = save;
}

void run_mamba(hipStream_t s, Arena& ar, const MambaP& mp,
               const float* xtok, const float* extra, float* out)
{
  size_t save = ar.off;
  float* xz = (float*)ar.alloc(S128B);
  k_gemm<2,8,0><<<128,256,0,s>>>(xtok, 64, mp.in_w, 128, nullptr, xz, 128);
  float* xc = (float*)ar.alloc(S64B);
  k_convsilu<<<G64,256,0,s>>>(xz, 128, mp.conv_w, mp.conv_b, xc);
  const float* src_f = xc;
  float* e = nullptr;
  if (mp.cross){
    e = (float*)ar.alloc(S64B);
    k_gemm<2,4,0><<<128,256,0,s>>>(extra, 64, mp.extra_w, 64, nullptr, e, 64);
    float* sf = (float*)ar.alloc(S64B);
    k_convsilu<<<G64,256,0,s>>>(e, 64, mp.econv_w, mp.econv_b, sf);
    src_f = sf;
  }
  float* y_f = (float*)ar.alloc(S64B);
  run_ssm(s, ar, src_f, xc, mp.x_w, mp.dt_w, mp.dt_b, mp.A_log, mp.D, y_f);

  float* xs_r = (float*)ar.alloc(S64B);
  k_reverse<<<G64,256,0,s>>>(xz, 128, xs_r);
  float* xc_r = (float*)ar.alloc(S64B);
  k_convsilu<<<G64,256,0,s>>>(xs_r, 64, mp.conv_w_b, mp.conv_b_b, xc_r);
  const float* src_b = xc_r;
  if (mp.cross){
    float* er = (float*)ar.alloc(S64B);
    k_reverse<<<G64,256,0,s>>>(e, 64, er);
    float* sb = (float*)ar.alloc(S64B);
    k_convsilu<<<G64,256,0,s>>>(er, 64, mp.econv_w_b, mp.econv_b_b, sb);
    src_b = sb;
  }
  float* y_b = (float*)ar.alloc(S64B);
  run_ssm(s, ar, src_b, xc_r, mp.x_w_b, mp.dt_w_b, mp.dt_b_b, mp.A_log_b, mp.D_b, y_b);

  float* y  = xs_r;  // reuse
  k_avgrev<<<G64,256,0,s>>>(y_f, y_b, y);
  float* y2 = y_f;   // reuse
  k_rmsgate<<<GLN,256,0,s>>>(y, mp.out_norm_w, xz + 64, 128, y2);
  k_gemm<2,4,0><<<128,256,0,s>>>(y2, 64, mp.out_w, 64, nullptr, out, 64);
  ar.off = save;
}

void run_cross(hipStream_t s, Arena& ar, const CrossP& cp,
               const float* x0, const float* x1, float* out)
{
  size_t save = ar.off;
  float* n0 = (float*)ar.alloc(S64B);
  k_layernorm<<<GLN,256,0,s>>>(x0, cp.ln0_w, cp.ln0_b, n0);
  float* n1 = (float*)ar.alloc(S64B);
  k_layernorm<<<GLN,256,0,s>>>(x1, cp.ln1_w, cp.ln1_b, n1);
  float* cat = (float*)ar.alloc(S128B);
  k_concat<<<G128,256,0,s>>>(n0, n1, cat);
  float* wgt = (float*)ar.alloc(S64B);
  k_gemm<4,4,2><<<128,256,0,s>>>(cat, 128, cp.cw_w, 64, cp.cw_b, wgt, 64);
  float* mo = (float*)ar.alloc(S64B);
  run_mamba(s, ar, cp.m, n0, n1, mo);
  k_crossgate<<<G64,256,0,s>>>(mo, wgt, n0, x0, out);
  ar.off = save;
}

} // namespace

extern "C" void kernel_launch(void* const* d_in, const int* in_sizes, int n_in,
                              void* d_out, int out_size, void* d_ws, size_t ws_size,
                              hipStream_t stream)
{
  (void)in_sizes; (void)out_size; (void)ws_size;
  if (n_in < 5) return;
  hipStream_t s = stream;

  // ---- parameters (jax pytree flatten: dict keys sorted, uppercase first)
  int pi = 4;
  auto nx = [&]{ return (const float*)d_in[pi++]; };
  const float* W1 = nx();       const float* W2 = nx();
  const float* alpha = nx();    const float* b1 = nx();
  const float* b2 = nx();       const float* beta = nx();
  const float* out_ln_b = nx(); const float* out_ln_w = nx();
  const float* self_ln_b = nx();const float* self_ln_w = nx();
  MambaP mself = readMamba(d_in, pi, false);
  CrossP spaC  = readCross(d_in, pi);
  CrossP spaL0 = readCross(d_in, pi);
  CrossP spaL1 = readCross(d_in, pi);
  CrossP speC  = readCross(d_in, pi);
  CrossP speL0 = readCross(d_in, pi);
  CrossP speL1 = readCross(d_in, pi);

  // ---- persistent buffers
  Arena ar{ (char*)d_ws, 0 };
  float* x1  = (float*)ar.alloc(S64B);
  float* x2  = (float*)ar.alloc(S64B);
  float* s1  = (float*)ar.alloc(S64B);
  float* s2  = (float*)ar.alloc(S64B);
  float* spaF= (float*)ar.alloc(S64B);
  float* speF= (float*)ar.alloc(S64B);
  float* fus = (float*)ar.alloc(S64B);
  float* lnB = (float*)ar.alloc(S64B);
  float* f2  = (float*)ar.alloc(S64B);
  float* h1  = (float*)ar.alloc(S128B);

  dim3 tb(32,8), tg(FM_L/32, FM_D/32, FM_B);
  k_tok<<<tg,tb,0,s>>>((const float*)d_in[0], x1);
  k_tok<<<tg,tb,0,s>>>((const float*)d_in[1], x2);
  k_tok<<<tg,tb,0,s>>>((const float*)d_in[2], s1);
  k_tok<<<tg,tb,0,s>>>((const float*)d_in[3], s2);

  // ---- layer stack (crossgate is index-aligned -> in-place is safe)
  run_cross(s, ar, spaL0, x1, s1, x1);
  run_cross(s, ar, speL0, x2, s2, x2);
  run_cross(s, ar, spaL1, x1, s1, x1);
  run_cross(s, ar, speL1, x2, s2, x2);

  run_cross(s, ar, spaC, x1, x2, spaF);
  run_cross(s, ar, speC, x2, x1, speF);
  k_fuse<<<G64,256,0,s>>>(spaF, speF, alpha, beta, fus);

  // ---- self block (single_block)
  {
    size_t save = ar.off;
    float* n  = (float*)ar.alloc(S64B);
    k_layernorm<<<GLN,256,0,s>>>(fus, self_ln_w, self_ln_b, n);
    float* mo = (float*)ar.alloc(S64B);
    run_mamba(s, ar, mself, n, nullptr, mo);
    k_add<<<G64,256,0,s>>>(mo, fus, fus);
    ar.off = save;
  }

  // ---- output head
  k_layernorm<<<GLN,256,0,s>>>(fus, out_ln_w, out_ln_b, lnB);
  k_gemm<2,8,3><<<128,256,0,s>>>(lnB, 64, W1, 128, b1, h1, 128);   // gelu
  k_gemm<4,4,0><<<128,256,0,s>>>(h1, 128, W2, 64, b2, f2, 64);
  k_final<<<tg,tb,0,s>>>(f2, x1, x2, (float*)d_out);
}
